// ScalarDotProductCriticNetwork_78649441124610
// MI455X (gfx1250) — matrix operand order repaired and checked by reading, not compiled, not run on hardware
//
#include <hip/hip_runtime.h>
#include <math.h>

typedef __bf16 bf16_t;
typedef __attribute__((ext_vector_type(16))) __bf16 v16bf;
typedef __attribute__((ext_vector_type(8)))  float  v8f;

#define N_AG   32
#define OBS    128
#define ACT    16
#define D_K    128
#define D_V    128
#define D_OA   144
#define D_OA_P 160
#define H_DIM  64

#define WS_WK 0
#define WS_WQ 16384
#define WS_WV 32768
#define WS_W1 53248
#define WS_TOTAL 61440

static __device__ __forceinline__ bf16_t to_bf(float x) { return (bf16_t)x; }

// A: 16x32 (MxK) fragment from row-major source, row stride ldk.
static __device__ __forceinline__ v16bf load_a_frag(const bf16_t* A, int m_base, int k_base,
                                                    int ldk, int lane) {
  const int m    = m_base + (lane & 15);
  const int half = (lane >> 4) & 1;
  const bf16_t* rowp = A + m * ldk + k_base;
  v16bf a;
#pragma unroll
  for (int v = 0; v < 8; ++v) {
    int k = ((v >> 2) << 4) + (half << 3) + ((v & 3) << 1);
    a[2 * v]     = rowp[k];
    a[2 * v + 1] = rowp[k + 1];
  }
  return a;
}

// B: 32x16 (KxN) fragment; source stored [n][k] row-major (== WMMA-B column-major).
static __device__ __forceinline__ v16bf load_b_frag(const bf16_t* Bcm, int n_base, int k_base,
                                                    int ldk, int lane) {
  const int n    = n_base + (lane & 15);
  const int half = (lane >> 4) & 1;
  const bf16_t* colp = Bcm + n * ldk + k_base;
  v16bf b;
#pragma unroll
  for (int v = 0; v < 8; ++v) {
    int k = ((v >> 2) << 4) + (half << 3) + ((v & 3) << 1);
    b[2 * v]     = colp[k];
    b[2 * v + 1] = colp[k + 1];
  }
  return b;
}

static __device__ __forceinline__ v8f wmma_bf16(v16bf a, v16bf b, v8f c) {
  return __builtin_amdgcn_wmma_f32_16x16x32_bf16(false, a, false, b, (short)0, c, false, false);
}

__global__ void prep_weights(const float* __restrict__ Wk, const float* __restrict__ Wq,
                             const float* __restrict__ Wv, const float* __restrict__ W1,
                             bf16_t* __restrict__ ws) {
  int idx = blockIdx.x * blockDim.x + threadIdx.x;
  if (idx < 16384) {
    ws[WS_WK + idx] = to_bf(Wk[idx]);
  } else if (idx < 32768) {
    ws[idx] = to_bf(Wq[idx - 16384]);
  } else if (idx < 53248) {
    int t = idx - 32768;
    int e = t / D_OA_P, c = t % D_OA_P;
    ws[idx] = to_bf(c < D_OA ? Wv[e * D_OA + c] : 0.0f);
  } else if (idx < WS_TOTAL) {
    ws[idx] = to_bf(W1[idx - 53248]);
  }
}

__global__ __launch_bounds__(256)
void critic_fused(const float* __restrict__ states, const float* __restrict__ policies,
                  const float* __restrict__ actions, const float* __restrict__ W2,
                  const bf16_t* __restrict__ ws,
                  float* __restrict__ out_value, float* __restrict__ out_weight) {
  __shared__ __align__(16) unsigned char smem[47360];
  bf16_t* OA     = (bf16_t*)(smem);
  bf16_t* OP     = OA + N_AG * D_OA_P;
  bf16_t* Qbf    = (bf16_t*)(smem + 24576);
  bf16_t* Kbf    = Qbf + N_AG * D_K;
  bf16_t* avT    = (bf16_t*)(smem + 24576);
  bf16_t* dlt    = (bf16_t*)(smem + 24576 + 8192);
  bf16_t* basebf = (bf16_t*)(smem);
  float*  hb     = (float*)(smem + 8192);
  float*  hd     = (float*)(smem + 16384);
  float*  sw     = (float*)(smem + 40960);
  bf16_t* wbf    = (bf16_t*)(smem + 45056);
  float*  w2s    = (float*)(smem + 47104);

  const int bid  = blockIdx.x;
  const int tid  = threadIdx.x;
  const int wave = tid >> 5;
  const int lane = tid & 31;
  const int lh   = lane >> 4;
  const int ln   = lane & 15;

  for (int idx = tid; idx < N_AG * D_OA_P; idx += 256) {
    int j = idx / D_OA_P, c = idx % D_OA_P;
    float va = 0.f, vp = 0.f;
    if (c < OBS) {
      va = vp = states[(bid * N_AG + j) * OBS + c];
    } else if (c < D_OA) {
      va = actions[(bid * N_AG + j) * ACT + (c - OBS)];
      vp = policies[(bid * N_AG + j) * ACT + (c - OBS)];
    }
    OA[idx] = to_bf(va);
    OP[idx] = to_bf(vp);
  }
  if (tid < H_DIM) w2s[tid] = W2[tid];
  __syncthreads();

  for (int r = 0; r < 2; ++r) {
    int pos = wave + (r << 3);
    int mt = pos >> 3, nt = pos & 7;
    v8f cq = {}, ck = {};
#pragma unroll
    for (int ks = 0; ks < 4; ++ks) {
      v16bf a  = load_a_frag(OA, mt * 16, ks * 32, D_OA_P, lane);
      v16bf bq = load_b_frag(ws + WS_WQ, nt * 16, ks * 32, OBS, lane);
      v16bf bk = load_b_frag(ws + WS_WK, nt * 16, ks * 32, OBS, lane);
      cq = wmma_bf16(a, bq, cq);
      ck = wmma_bf16(a, bk, ck);
    }
#pragma unroll
    for (int v = 0; v < 8; ++v) {
      int m = mt * 16 + v + (lh << 3), n = nt * 16 + ln;
      Qbf[m * D_K + n] = to_bf(cq[v]);
      Kbf[m * D_K + n] = to_bf(ck[v]);
    }
  }
  __syncthreads();

  if (wave < 4) {
    int mt = wave >> 1, nt = wave & 1;
    v8f c = {};
#pragma unroll
    for (int ks = 0; ks < 4; ++ks) {
      v16bf a = load_a_frag(Qbf, mt * 16, ks * 32, D_K, lane);
      v16bf b = load_b_frag(Kbf, nt * 16, ks * 32, D_K, lane);
      c = wmma_bf16(a, b, c);
    }
    const float scale = 0.08838834764831845f;
#pragma unroll
    for (int v = 0; v < 8; ++v)
      sw[(mt * 16 + v + (lh << 3)) * N_AG + nt * 16 + ln] = c[v] * scale;
  }
  __syncthreads();

  if (tid < N_AG) {
    float mx = -1e30f;
    for (int j = 0; j < N_AG; ++j) mx = fmaxf(mx, sw[tid * N_AG + j]);
    float sum = 0.f;
    float e[N_AG];
    for (int j = 0; j < N_AG; ++j) { e[j] = __expf(sw[tid * N_AG + j] - mx); sum += e[j]; }
    float inv = 1.f / sum;
    for (int j = 0; j < N_AG; ++j) {
      float wv = e[j] * inv;
      sw[tid * N_AG + j]  = wv;
      wbf[tid * N_AG + j] = to_bf(wv);
      out_weight[bid * N_AG * N_AG + tid * N_AG + j] = wv;
    }
  }
  __syncthreads();

  for (int r = 0; r < 2; ++r) {
    int pos = wave + (r << 3);
    int mt = pos >> 3, nt = pos & 7;
    v8f ca = {}, cp = {};
#pragma unroll
    for (int ks = 0; ks < 5; ++ks) {
      v16bf aa = load_a_frag(OA, mt * 16, ks * 32, D_OA_P, lane);
      v16bf ap = load_a_frag(OP, mt * 16, ks * 32, D_OA_P, lane);
      v16bf b  = load_b_frag(ws + WS_WV, nt * 16, ks * 32, D_OA_P, lane);
      ca = wmma_bf16(aa, b, ca);
      cp = wmma_bf16(ap, b, cp);
    }
#pragma unroll
    for (int v = 0; v < 8; ++v) {
      int k = mt * 16 + v + (lh << 3);
      int e = nt * 16 + ln;
      float ta = tanhf(ca[v]);
      float tp = tanhf(cp[v]);
      avT[e * N_AG + k] = to_bf(ta);
      dlt[k * D_V + e]  = to_bf(tp - ta);
    }
  }
  __syncthreads();

  for (int r = 0; r < 2; ++r) {
    int pos = wave + (r << 3);
    int mt = pos >> 3, nt = pos & 7;
    v16bf a = load_a_frag(wbf, mt * 16, 0, N_AG, lane);
    v16bf b = load_b_frag(avT, nt * 16, 0, N_AG, lane);
    v8f c = {};
    c = wmma_bf16(a, b, c);
#pragma unroll
    for (int v = 0; v < 8; ++v)
      basebf[(mt * 16 + v + (lh << 3)) * D_V + nt * 16 + ln] = to_bf(c[v]);
  }
  __syncthreads();

  for (int r = 0; r < 2; ++r) {
    int t = wave + (r << 3);
    const bf16_t* Amat = (t < 8) ? basebf : dlt;
    float* H = (t < 8) ? hb : hd;
    int p = t & 7, mt = p >> 2, nt = p & 3;
    v8f c = {};
#pragma unroll
    for (int ks = 0; ks < 4; ++ks) {
      v16bf a = load_a_frag(Amat, mt * 16, ks * 32, D_V, lane);
      v16bf b = load_b_frag(ws + WS_W1, nt * 16, ks * 32, D_V, lane);
      c = wmma_bf16(a, b, c);
    }
#pragma unroll
    for (int v = 0; v < 8; ++v)
      H[(mt * 16 + v + (lh << 3)) * H_DIM + nt * 16 + ln] = c[v];
  }
  __syncthreads();

  for (int pair = tid; pair < N_AG * N_AG; pair += 256) {
    int i = pair >> 5, j = pair & 31;
    float wij = sw[i * N_AG + j];
    float acc = 0.f;
#pragma unroll 8
    for (int h = 0; h < H_DIM; ++h) {
      float x = hb[i * H_DIM + h] + wij * hd[j * H_DIM + h];
      x = (x > 0.f) ? x : 0.01f * x;
      acc = fmaf(w2s[h], x, acc);
    }
    out_value[bid * N_AG * N_AG + pair] = acc;
  }
}

extern "C" void kernel_launch(void* const* d_in, const int* in_sizes, int n_in,
                              void* d_out, int out_size, void* d_ws, size_t ws_size,
                              hipStream_t stream) {
  const float* states   = (const float*)d_in[0];
  const float* policies = (const float*)d_in[1];
  const float* actions  = (const float*)d_in[2];
  const float* Wk       = (const float*)d_in[3];
  const float* Wq       = (const float*)d_in[4];
  const float* Wv       = (const float*)d_in[5];
  const float* W1       = (const float*)d_in[6];
  const float* W2       = (const float*)d_in[7];
  bf16_t* ws = (bf16_t*)d_ws;
  float* out = (float*)d_out;
  float* out_value  = out;
  float* out_weight = out + 128 * 32 * 32;

  prep_weights<<<(WS_TOTAL + 255) / 256, 256, 0, stream>>>(Wk, Wq, Wv, W1, ws);
  critic_fused<<<128, 256, 0, stream>>>(states, policies, actions, W2, ws, out_value, out_weight);
}
